// BEVTrainer_75024488726788
// MI455X (gfx1250) — matrix operand order repaired and checked by reading, not compiled, not run on hardware
//
#include <hip/hip_runtime.h>
#include <hip/hip_bf16.h>
#include <math.h>

// MI455X / gfx1250, wave32, WMMA f32_16x16x32_f16 GEMM pipeline.
// Workspace requirement: ~113 MB (buffers reuse scratch regions sequentially).

typedef __attribute__((ext_vector_type(16))) _Float16 v16h;
typedef __attribute__((ext_vector_type(8)))  float    v8f;

namespace {
constexpr int kB = 2, kC = 6, kNPB = 20000, kD = 64, kS = 4096, kH = 28, kW = 50;
constexpr int kN   = kB * kNPB;   // 40000
constexpr int kBC  = kB * kC;     // 12
constexpr int kHW  = kH * kW;     // 1400
constexpr int kHWP = 1408;        // padded to mult of 16/32/64
constexpr int kM   = 2048;
constexpr int kNPAIR = kBC * 5000;
constexpr float kScaleInv = 0.125f;   // 1/sqrt(64)
constexpr float kTempInv  = 1.0f / 0.07f;
}

__device__ __forceinline__ int lane_id() { return (int)(threadIdx.x & 31u); }

__device__ __forceinline__ v8f wmma_f32_f16(v16h a, v16h b, v8f c) {
  return __builtin_amdgcn_wmma_f32_16x16x32_f16(false, a, false, b, (short)0, c, false, false);
}

// ---- fragment loaders (wave32 WMMA; A and B share a contiguous k order, not the ISA table order) ----
// A (16x32 f16): lane holds row M=lane&15; half i -> K = (lane>>4)*16 + i
__device__ __forceinline__ v16h frag_a_f32row(const float* __restrict__ A, int lda, int m0, int k0) {
  const int l = lane_id(); const int r = m0 + (l & 15); const int hi = l >> 4;
  const float* p = A + (size_t)r * lda + k0 + hi * 16;
  v16h f;
#pragma unroll
  for (int i = 0; i < 16; ++i) f[i] = (_Float16)p[i];
  return f;
}
__device__ __forceinline__ v16h frag_a_f16row(const _Float16* __restrict__ A, int lda, int m0, int k0) {
  const int l = lane_id(); const int r = m0 + (l & 15); const int hi = l >> 4;
  const _Float16* p = A + (size_t)r * lda + k0 + hi * 16;
  v16h f;
#pragma unroll
  for (int i = 0; i < 16; ++i) f[i] = p[i];
  return f;
}
// A from column-strided f32 (A[m][k] = src[k*ld + m]); branch-free row guard:
// clamped (in-bounds) address + value select, so no exec-mask branches.
__device__ __forceinline__ v16h frag_a_f32col(const float* __restrict__ A, int ld, int m0, int k0, int mmax) {
  const int l = lane_id(); const int r = m0 + (l & 15); const int hi = l >> 4;
  const bool ok = r < mmax;
  const int rr = ok ? r : 0;
  v16h f;
#pragma unroll
  for (int i = 0; i < 16; ++i) {
    int k = k0 + hi * 16 + i;
    float v = A[(size_t)k * ld + rr];
    f[i] = ok ? (_Float16)v : (_Float16)0.f;
  }
  return f;
}
// B (32x16 f16): lane holds col N=lane&15; half i -> K = (lane>>4)*16 + i
// trans: B[k][n] = src[n*ldb + k] (contiguous per lane)
__device__ __forceinline__ v16h frag_b_f32trans(const float* __restrict__ Bp, int ldb, int n0, int k0) {
  const int l = lane_id(); const int n = n0 + (l & 15); const int kb = k0 + (l >> 4) * 16;
  const float* p = Bp + (size_t)n * ldb + kb;
  v16h f;
#pragma unroll
  for (int i = 0; i < 16; ++i) f[i] = (_Float16)p[i];
  return f;
}
__device__ __forceinline__ v16h frag_b_f16trans(const _Float16* __restrict__ Bp, int ldb, int n0, int k0) {
  const int l = lane_id(); const int n = n0 + (l & 15); const int kb = k0 + (l >> 4) * 16;
  const _Float16* p = Bp + (size_t)n * ldb + kb;
  v16h f;
#pragma unroll
  for (int i = 0; i < 16; ++i) f[i] = p[i];
  return f;
}
// rowmajor: B[k][n] = src[k*ldb + n]; branch-free col guard (clamp + select)
__device__ __forceinline__ v16h frag_b_f32row(const float* __restrict__ Bp, int ldb, int n0, int k0, int nmax) {
  const int l = lane_id(); const int n = n0 + (l & 15); const int kb = k0 + (l >> 4) * 16;
  const bool ok = n < nmax;
  const int nn = ok ? n : 0;
  v16h f;
#pragma unroll
  for (int i = 0; i < 16; ++i) {
    float v = Bp[(size_t)(kb + i) * ldb + nn];
    f[i] = ok ? (_Float16)v : (_Float16)0.f;
  }
  return f;
}

// ---- block reductions (deterministic tree) ----
template <int BS>
__device__ __forceinline__ float block_sum(float v) {
  __shared__ float sh[BS];
  int t = threadIdx.x;
  sh[t] = v; __syncthreads();
#pragma unroll
  for (int s = BS / 2; s > 0; s >>= 1) { if (t < s) sh[t] += sh[t + s]; __syncthreads(); }
  float r = sh[0]; __syncthreads();
  return r;
}
template <int BS>
__device__ __forceinline__ float block_max(float v) {
  __shared__ float sh[BS];
  int t = threadIdx.x;
  sh[t] = v; __syncthreads();
#pragma unroll
  for (int s = BS / 2; s > 0; s >>= 1) { if (t < s) sh[t] = fmaxf(sh[t], sh[t + s]); __syncthreads(); }
  float r = sh[0]; __syncthreads();
  return r;
}

// ---- setup kernels ----
__global__ void k_zero8(unsigned char* p, long n) {
  long i = (long)blockIdx.x * blockDim.x + threadIdx.x;
  if (i < n) p[i] = 0;
}
__global__ void k_scatter_mask(const int* __restrict__ pimg, const int* __restrict__ ppts,
                               unsigned char* __restrict__ mask, int n) {
  int i = blockIdx.x * blockDim.x + threadIdx.x;
  if (i < n) mask[(size_t)pimg[3 * i] * kN + ppts[i]] = 1;
}
__global__ void k_gather_feat(const float* __restrict__ vox, const int* __restrict__ sidx,
                              float* __restrict__ samp) {
  long i = (long)blockIdx.x * blockDim.x + threadIdx.x;
  const long total = (long)kBC * kS * kD;
  if (i >= total) return;
  int c = (int)(i & 63);
  long bs = i >> 6;
  int b = (int)(bs / kS);
  int s = (int)(bs - (long)b * kS);
  int g = (b / kC) * kNPB + sidx[b * kS + s];
  samp[i] = vox[(size_t)g * kD + c];
}
__global__ void k_geom(const float* __restrict__ coords, const int* __restrict__ sidx,
                       const float* __restrict__ Rd, const float* __restrict__ Td,
                       const float* __restrict__ Kd, const unsigned char* __restrict__ mask,
                       float* __restrict__ uvbuf, float* __restrict__ mskbuf) {
  int i = blockIdx.x * blockDim.x + threadIdx.x;
  if (i >= kBC * kS) return;
  int b = i / kS, s = i - b * kS;
  int g = (b / kC) * kNPB + sidx[b * kS + s];
  float x = coords[3 * g], y = coords[3 * g + 1], z = coords[3 * g + 2];
  const float* R = Rd + 9 * b; const float* T = Td + 3 * b; const float* K = Kd + 9 * b;
  float cx = R[0] * x + R[1] * y + R[2] * z + T[0];
  float cy = R[3] * x + R[4] * y + R[5] * z + T[1];
  float cz = R[6] * x + R[7] * y + R[8] * z + T[2];
  float px = K[0] * cx + K[1] * cy + K[2] * cz;
  float py = K[3] * cx + K[4] * cy + K[5] * cz;
  float pz = K[6] * cx + K[7] * cy + K[8] * cz;
  uvbuf[2 * i] = px / pz;
  uvbuf[2 * i + 1] = py / pz;
  mskbuf[i] = mask[(size_t)b * kN + g] ? 1.f : 0.f;
}

// ---- WMMA GEMM kernels (4 waves/block, 4 accumulators per wave) ----
// C(64 x L) = W(64x64) * B. Each wave: one 16-col n-tile x full 64 rows,
// B fragment loaded once per k-step (4x reuse).
// BTRANS: B[k][n]=bsrc[n*ldb+k] else B[k][n]=bsrc[k*ldb+n]
template <bool BTRANS>
__global__ void k_gemm_w(const float* __restrict__ Wm, const float* __restrict__ Bsrc,
                         float* __restrict__ Cdst, int L, int ldb, long strideB, long strideC) {
  const int ntiles = (L + 15) >> 4;
  const int nt = blockIdx.x * 4 + (threadIdx.x >> 5);
  if (nt >= ntiles) return;
  const int n0 = nt * 16;
  const float* bp = Bsrc + (size_t)blockIdx.z * strideB;
  float* cp = Cdst + (size_t)blockIdx.z * strideC;
  v8f acc[4];
#pragma unroll
  for (int j = 0; j < 4; ++j) acc[j] = (v8f){0, 0, 0, 0, 0, 0, 0, 0};
#pragma unroll
  for (int k0 = 0; k0 < 64; k0 += 32) {
    v16h b;
    if (BTRANS) b = frag_b_f32trans(bp, ldb, n0, k0);
    else        b = frag_b_f32row(bp, ldb, n0, k0, L);
#pragma unroll
    for (int j = 0; j < 4; ++j) {
      v16h a = frag_a_f32row(Wm, 64, j * 16, k0);
      acc[j] = wmma_f32_f16(a, b, acc[j]);
    }
  }
  const int l = lane_id(); const int hi = l >> 4; const int n = n0 + (l & 15);
  if (n < L) {
#pragma unroll
    for (int j = 0; j < 4; ++j)
#pragma unroll
      for (int r = 0; r < 8; ++r)
        cp[(size_t)(j * 16 + r + 8 * hi) * L + n] = acc[j][r];
  }
}

// Out = Src^T(L x 64) * W^T + bias (f16). Each wave: one 16-row m-tile x all 64
// output cols; A fragment loaded once per k-step. TRANSOUT=false: Out[m*64+n]
// (Lp x 64); TRANSOUT=true: Out[n*Lp+m] (64 x Lp). Pad rows (m>=L) written 0.
template <bool TRANSOUT>
__global__ void k_proj(const float* __restrict__ Src, int L, int Lp,
                       const float* __restrict__ Wm, const float* __restrict__ bias,
                       _Float16* __restrict__ Out, long strideS, long strideO) {
  const int mtiles = Lp >> 4;
  const int mt = blockIdx.x * 4 + (threadIdx.x >> 5);
  if (mt >= mtiles) return;
  const int m0 = mt * 16;
  const float* sp = Src + (size_t)blockIdx.z * strideS;
  _Float16* op = Out + (size_t)blockIdx.z * strideO;
  v8f acc[4];
#pragma unroll
  for (int j = 0; j < 4; ++j) acc[j] = (v8f){0, 0, 0, 0, 0, 0, 0, 0};
#pragma unroll
  for (int k0 = 0; k0 < 64; k0 += 32) {
    v16h a = frag_a_f32col(sp, L, m0, k0, L);
#pragma unroll
    for (int j = 0; j < 4; ++j) {
      v16h b = frag_b_f32trans(Wm, 64, j * 16, k0);
      acc[j] = wmma_f32_f16(a, b, acc[j]);
    }
  }
  const int l = lane_id(); const int hi = l >> 4;
#pragma unroll
  for (int j = 0; j < 4; ++j) {
    const int n = j * 16 + (l & 15);
    const float bv = bias[n];
#pragma unroll
    for (int r = 0; r < 8; ++r) {
      int m = m0 + r + 8 * hi;
      float v = (m < L) ? acc[j][r] + bv : 0.f;
      if (TRANSOUT) op[(size_t)n * Lp + m] = (_Float16)v;
      else          op[(size_t)m * 64 + n] = (_Float16)v;
    }
  }
}

// C[m][n] = scale * sum_k A[m*64+k] * B[n*64+k]. Each wave: 16 rows x 64 cols
// (4 n-tiles), A loaded once per k-step. Np must be a multiple of 64.
__global__ void k_gemm_f16(const _Float16* __restrict__ A, const _Float16* __restrict__ Bm,
                           float* __restrict__ Cdst, int Np, float scale) {
  const int ngroups = Np >> 6;
  const int g = blockIdx.x * 4 + (threadIdx.x >> 5);
  if (g >= ngroups) return;
  const int m0 = blockIdx.y * 16, n0 = g * 64;
  v8f acc[4];
#pragma unroll
  for (int j = 0; j < 4; ++j) acc[j] = (v8f){0, 0, 0, 0, 0, 0, 0, 0};
#pragma unroll
  for (int k0 = 0; k0 < 64; k0 += 32) {
    v16h a = frag_a_f16row(A, 64, m0, k0);
#pragma unroll
    for (int j = 0; j < 4; ++j)
      acc[j] = wmma_f32_f16(a, frag_b_f16trans(Bm, 64, n0 + j * 16, k0), acc[j]);
  }
  const int l = lane_id(); const int hi = l >> 4;
#pragma unroll
  for (int j = 0; j < 4; ++j) {
    const int n = n0 + j * 16 + (l & 15);
#pragma unroll
    for (int r = 0; r < 8; ++r)
      Cdst[(size_t)(m0 + r + 8 * hi) * Np + n] = scale * acc[j][r];
  }
}

// Fused[d*L+q] = Base[d*L+q] + sum_k Sc[q*Kp+k] * Vt[d*Kp+k]
// Vt is the transposed V (64 x Kp, f16) so B fragments are contiguous.
// Each wave: one 16-row q-tile x all 64 d; A loaded once per k-step.
__global__ void k_av_add(const float* __restrict__ Sc, int Kp, const _Float16* __restrict__ Vt,
                         const float* __restrict__ Base, float* __restrict__ Fused, int L) {
  const int mtiles = (L + 15) >> 4;
  const int mt = blockIdx.x * 4 + (threadIdx.x >> 5);
  if (mt >= mtiles) return;
  const int m0 = mt * 16;
  v8f acc[4];
#pragma unroll
  for (int j = 0; j < 4; ++j) acc[j] = (v8f){0, 0, 0, 0, 0, 0, 0, 0};
  for (int k0 = 0; k0 < Kp; k0 += 32) {
    v16h a = frag_a_f32row(Sc, Kp, m0, k0);
#pragma unroll
    for (int j = 0; j < 4; ++j)
      acc[j] = wmma_f32_f16(a, frag_b_f16trans(Vt, Kp, j * 16, k0), acc[j]);
  }
  const int l = lane_id(); const int hi = l >> 4;
#pragma unroll
  for (int j = 0; j < 4; ++j) {
    const int d = j * 16 + (l & 15);
#pragma unroll
    for (int r = 0; r < 8; ++r) {
      int q = m0 + r + 8 * hi;
      if (q < L) {
        size_t off = (size_t)d * L + q;
        Fused[off] = Base[off] + acc[j][r];
      }
    }
  }
}

// ---- softmax / contrastive / BN / reductions ----
__global__ void k_softmax(float* __restrict__ Sc, int cols, int ld) {
  float* p = Sc + (size_t)blockIdx.x * ld;
  int t = threadIdx.x;
  float mx = -3.402823466e38f;
  for (int c = t; c < cols; c += 256) mx = fmaxf(mx, p[c]);
  mx = block_max<256>(mx);
  float s = 0.f;
  for (int c = t; c < cols; c += 256) { float e = __expf(p[c] - mx); p[c] = e; s += e; }
  s = block_sum<256>(s);
  float inv = 1.f / s;
  for (int c = t; c < cols; c += 256) p[c] *= inv;
  for (int c = cols + t; c < ld; c += 256) p[c] = 0.f;  // zero pad cols for AV GEMM
}

__global__ void k_extract_norm(const float* __restrict__ Fused, long strideF, int ld,
                               const int* __restrict__ Idx, int idxStride,
                               _Float16* __restrict__ Out) {
  int m = blockIdx.x, t = threadIdx.x;  // blockDim = 64
  const int* row = Idx + m * idxStride;
  int bb = row[0];
  int col = (idxStride == 2) ? row[1] : (row[1] * kW + row[2]);
  float v = Fused[(size_t)bb * strideF + (size_t)t * ld + col];
  __shared__ float sh[64];
  sh[t] = v * v; __syncthreads();
  for (int s2 = 32; s2 > 0; s2 >>= 1) { if (t < s2) sh[t] += sh[t + s2]; __syncthreads(); }
  float nrm = fmaxf(sqrtf(sh[0]), 1e-12f);
  Out[m * 64 + t] = (_Float16)(v / nrm);
}

__global__ void k_lse(const float* __restrict__ Lg, float* __restrict__ contrib) {
  const float* p = Lg + (size_t)blockIdx.x * kM;
  int t = threadIdx.x;
  float mx = -3.402823466e38f;
  for (int c = t; c < kM; c += 256) mx = fmaxf(mx, p[c]);
  mx = block_max<256>(mx);
  float s = 0.f;
  for (int c = t; c < kM; c += 256) s += __expf(p[c] - mx);
  s = block_sum<256>(s);
  if (t == 0) contrib[blockIdx.x] = mx + __logf(s) - p[blockIdx.x];
}

__global__ void k_bnstats(const float* __restrict__ Z, int L, long bstride,
                          float* __restrict__ meanOut, float* __restrict__ invOut) {
  int ch = blockIdx.x; int t = threadIdx.x;
  float s = 0.f, s2 = 0.f;
  for (int b = 0; b < kBC; ++b) {
    const float* p = Z + (size_t)b * bstride + (size_t)ch * L;
    for (int i = t; i < L; i += 256) { float x = p[i]; s += x; s2 += x * x; }
  }
  s = block_sum<256>(s);
  s2 = block_sum<256>(s2);
  if (t == 0) {
    float cnt = (float)kBC * (float)L;
    float m = s / cnt;
    float var = s2 / cnt - m * m;
    meanOut[ch] = m;
    invOut[ch] = rsqrtf(var + 1e-5f);
  }
}

__global__ void k_bnrelu(float* __restrict__ Z, int L, const float* __restrict__ mean,
                         const float* __restrict__ inv, const float* __restrict__ g,
                         const float* __restrict__ bb, long total) {
  long i = (long)blockIdx.x * blockDim.x + threadIdx.x;
  if (i >= total) return;
  int ch = (int)((i / L) & 63);
  float x = (Z[i] - mean[ch]) * inv[ch] * g[ch] + bb[ch];
  Z[i] = fmaxf(x, 0.f);
}

__global__ void k_headscore(const float* __restrict__ Z, int L, long bstride,
                            const float* __restrict__ W3, float* __restrict__ outb) {
  long i = (long)blockIdx.x * blockDim.x + threadIdx.x;
  long total = (long)kBC * L;
  if (i >= total) return;
  int b = (int)(i / L); int l = (int)(i - (long)b * L);
  const float* p = Z + (size_t)b * bstride + l;
  float s = 0.f;
#pragma unroll
  for (int c = 0; c < 64; ++c) s += W3[c] * p[(size_t)c * L];
  outb[i] = 1.f / (1.f + __expf(-s));
}

__global__ void k_reduce(const float* __restrict__ src, long n, float* __restrict__ dst) {
  int t = threadIdx.x;
  float s = 0.f;
  for (long i = t; i < n; i += 256) s += src[i];
  s = block_sum<256>(s);
  if (t == 0) dst[0] = s;
}

__global__ void k_final(const float* __restrict__ hdr, float* __restrict__ out) {
  float loss = hdr[4] / (float)kM;
  float total = loss
      + hdr[2] / (float)(kBC * kS)
      + hdr[3] / (float)(kBC * kHW)
      + 0.001f * hdr[0] / (float)(kBC * kS * 2)
      + 0.001f * hdr[1] / (float)(kBC * kS);
  out[0] = total;
}

extern "C" void kernel_launch(void* const* d_in, const int* in_sizes, int n_in,
                              void* d_out, int out_size, void* d_ws, size_t ws_size,
                              hipStream_t stream) {
  (void)in_sizes; (void)n_in; (void)out_size; (void)ws_size;
  const float* vox    = (const float*)d_in[0];
  const float* coords = (const float*)d_in[1];
  const float* imgf   = (const float*)d_in[2];
  const float* Rd = (const float*)d_in[3];
  const float* Td = (const float*)d_in[4];
  const float* Kd = (const float*)d_in[5];
  const float* W_pc   = (const float*)d_in[6];
  const float* W_imgc = (const float*)d_in[7];
  const float* Wq = (const float*)d_in[8];  const float* bq = (const float*)d_in[9];
  const float* Wk = (const float*)d_in[10]; const float* bk = (const float*)d_in[11];
  const float* Wv = (const float*)d_in[12]; const float* bv = (const float*)d_in[13];
  const float* Wps1 = (const float*)d_in[14]; const float* gp1 = (const float*)d_in[15]; const float* bp1 = (const float*)d_in[16];
  const float* Wps2 = (const float*)d_in[17]; const float* gp2 = (const float*)d_in[18]; const float* bp2 = (const float*)d_in[19];
  const float* Wps3 = (const float*)d_in[20];
  const float* Wis1 = (const float*)d_in[21]; const float* gi1 = (const float*)d_in[22]; const float* bi1 = (const float*)d_in[23];
  const float* Wis2 = (const float*)d_in[24]; const float* gi2 = (const float*)d_in[25]; const float* bi2 = (const float*)d_in[26];
  const float* Wis3 = (const float*)d_in[27];
  const int* ppts = (const int*)d_in[28];
  const int* pimg = (const int*)d_in[29];
  const int* sidx = (const int*)d_in[30];
  const int* cpc  = (const int*)d_in[31];
  const int* cim  = (const int*)d_in[32];

  char* ws = (char*)d_ws;
  size_t off = 0;
  auto alloc = [&](size_t bytes) -> char* {
    off = (off + 255) & ~(size_t)255;
    char* p = ws + off;
    off += bytes;
    return p;
  };

  float* hdr           = (float*)alloc(1024 * sizeof(float));
  unsigned char* mask  = (unsigned char*)alloc((size_t)kBC * kN);
  float* samp          = (float*)alloc((size_t)kBC * kS * 64 * 4);
  float* pcfeat        = (float*)alloc((size_t)kBC * 64 * kS * 4);
  float* imgmid        = (float*)alloc((size_t)kBC * 64 * kHW * 4);
  _Float16* Qpc  = (_Float16*)alloc((size_t)kBC * kS * 64 * 2);
  _Float16* Kpc  = (_Float16*)alloc((size_t)kBC * kS * 64 * 2);
  _Float16* Vpct = (_Float16*)alloc((size_t)kBC * 64 * kS * 2);    // transposed V (64 x S)
  _Float16* Qim  = (_Float16*)alloc((size_t)kBC * kHWP * 64 * 2);
  _Float16* Kim  = (_Float16*)alloc((size_t)kBC * kHWP * 64 * 2);
  _Float16* Vimt = (_Float16*)alloc((size_t)kBC * 64 * kHWP * 2);  // transposed V (64 x HWP)
  float* scores = (float*)alloc((size_t)kS * kHWP * 4);     // reused per batch + both attns
  float* fpc    = (float*)alloc((size_t)kBC * 64 * kS * 4);
  float* fim    = (float*)alloc((size_t)kBC * 64 * kHW * 4);
  _Float16* pcc = (_Float16*)alloc((size_t)kM * 64 * 2);
  _Float16* imc = (_Float16*)alloc((size_t)kM * 64 * 2);
  float* logits  = (float*)alloc((size_t)kM * kM * 4);
  float* uvbuf   = (float*)alloc((size_t)kBC * kS * 2 * 4);
  float* mskbuf  = (float*)alloc((size_t)kBC * kS * 4);
  float* scbuf   = (float*)alloc((size_t)kBC * kS * 4);
  float* contrib = (float*)alloc((size_t)kM * 4);
  // region reuse (strictly sequential phases):
  float* zA  = scores;   // BN pc head scratch (12.6MB <= 23.1MB)
  float* zB  = pcfeat;   // pcfeat dead after attention loop
  float* zIA = samp;     // samp dead after pc_feat GEMM
  float* zIB = imgmid;   // imgmid dead after attention loop

  const long sPC  = (long)64 * kS;     // per-batch stride of (64 x S) maps
  const long sIM  = (long)64 * kHW;    // per-batch stride of (64 x HW) maps
  const long sQPC = (long)kS * 64;
  const long sQIM = (long)kHWP * 64;
  const long sVPT = (long)64 * kS;
  const long sVIT = (long)64 * kHWP;

  // 1) mask scatter + gathers + geometry
  k_zero8<<<dim3(((size_t)kBC * kN + 255) / 256), 256, 0, stream>>>(mask, (long)kBC * kN);
  k_scatter_mask<<<dim3((kNPAIR + 255) / 256), 256, 0, stream>>>(pimg, ppts, mask, kNPAIR);
  k_gather_feat<<<dim3(((long)kBC * kS * 64 + 255) / 256), 256, 0, stream>>>(vox, sidx, samp);
  k_geom<<<dim3((kBC * kS + 255) / 256), 256, 0, stream>>>(coords, sidx, Rd, Td, Kd, mask, uvbuf, mskbuf);

  // 2) pc_feat = W_pc @ samp^T ; img_mid = W_imgc @ img
  k_gemm_w<true ><<<dim3(64, 1, kBC), 128, 0, stream>>>(W_pc,   samp, pcfeat, kS,  64,  sQPC, sPC);
  k_gemm_w<false><<<dim3(22, 1, kBC), 128, 0, stream>>>(W_imgc, imgf, imgmid, kHW, kHW, sIM,  sIM);

  // 3) Q/K/V projections (f16; V stored transposed; pad rows zeroed)
  k_proj<false><<<dim3(64, 1, kBC), 128, 0, stream>>>(pcfeat, kS,  kS,   Wq, bq, Qpc,  sPC, sQPC);
  k_proj<false><<<dim3(64, 1, kBC), 128, 0, stream>>>(pcfeat, kS,  kS,   Wk, bk, Kpc,  sPC, sQPC);
  k_proj<true ><<<dim3(64, 1, kBC), 128, 0, stream>>>(pcfeat, kS,  kS,   Wv, bv, Vpct, sPC, sVPT);
  k_proj<false><<<dim3(22, 1, kBC), 128, 0, stream>>>(imgmid, kHW, kHWP, Wq, bq, Qim,  sIM, sQIM);
  k_proj<false><<<dim3(22, 1, kBC), 128, 0, stream>>>(imgmid, kHW, kHWP, Wk, bk, Kim,  sIM, sQIM);
  k_proj<true ><<<dim3(22, 1, kBC), 128, 0, stream>>>(imgmid, kHW, kHWP, Wv, bv, Vimt, sIM, sVIT);

  // 4) the two cross-attentions, batch-sequential (scores buffer reused)
  for (int b = 0; b < kBC; ++b) {
    const _Float16* qp  = Qpc  + (size_t)b * sQPC;
    const _Float16* kp  = Kpc  + (size_t)b * sQPC;
    const _Float16* vpt = Vpct + (size_t)b * sVPT;
    const _Float16* qi  = Qim  + (size_t)b * sQIM;
    const _Float16* ki  = Kim  + (size_t)b * sQIM;
    const _Float16* vit = Vimt + (size_t)b * sVIT;
    // attn1: Q=pc (S), K/V=img (HWP)
    k_gemm_f16<<<dim3(6, kS / 16, 1), 128, 0, stream>>>(qp, ki, scores, kHWP, kScaleInv);
    k_softmax<<<dim3(kS), 256, 0, stream>>>(scores, kHW, kHWP);
    k_av_add<<<dim3(64, 1, 1), 128, 0, stream>>>(scores, kHWP, vit,
        pcfeat + (size_t)b * sPC, fpc + (size_t)b * sPC, kS);
    // attn2: Q=img (HWP), K/V=pc (S)
    k_gemm_f16<<<dim3(16, kHWP / 16, 1), 128, 0, stream>>>(qi, kp, scores, kS, kScaleInv);
    k_softmax<<<dim3(kHW), 256, 0, stream>>>(scores, kS, kS);
    k_av_add<<<dim3(22, 1, 1), 128, 0, stream>>>(scores, kS, vpt,
        imgmid + (size_t)b * sIM, fim + (size_t)b * sIM, kHW);
  }

  // 5) contrastive loss
  k_extract_norm<<<dim3(kM), 64, 0, stream>>>(fpc, sPC, kS,  cpc, 2, pcc);
  k_extract_norm<<<dim3(kM), 64, 0, stream>>>(fim, sIM, kHW, cim, 3, imc);
  k_gemm_f16<<<dim3(8, kM / 16, 1), 128, 0, stream>>>(pcc, imc, logits, kM, kTempInv);
  k_lse<<<dim3(kM), 256, 0, stream>>>(logits, contrib);

  // 6) pc score head (BN over (b,l), per channel)
  k_gemm_w<false><<<dim3(64, 1, kBC), 128, 0, stream>>>(Wps1, fpc, zA, kS, kS, sPC, sPC);
  k_bnstats<<<dim3(64), 256, 0, stream>>>(zA, kS, sPC, hdr + 8, hdr + 72);
  k_bnrelu<<<dim3(12288), 256, 0, stream>>>(zA, kS, hdr + 8, hdr + 72, gp1, bp1, (long)kBC * 64 * kS);
  k_gemm_w<false><<<dim3(64, 1, kBC), 128, 0, stream>>>(Wps2, zA, zB, kS, kS, sPC, sPC);
  k_bnstats<<<dim3(64), 256, 0, stream>>>(zB, kS, sPC, hdr + 8, hdr + 72);
  k_bnrelu<<<dim3(12288), 256, 0, stream>>>(zB, kS, hdr + 8, hdr + 72, gp2, bp2, (long)kBC * 64 * kS);
  k_headscore<<<dim3(192), 256, 0, stream>>>(zB, kS, sPC, Wps3, scbuf);
  k_reduce<<<1, 256, 0, stream>>>(scbuf, (long)kBC * kS, hdr + 2);

  // 7) img score head
  k_gemm_w<false><<<dim3(22, 1, kBC), 128, 0, stream>>>(Wis1, fim, zIA, kHW, kHW, sIM, sIM);
  k_bnstats<<<dim3(64), 256, 0, stream>>>(zIA, kHW, sIM, hdr + 8, hdr + 72);
  k_bnrelu<<<dim3(4200), 256, 0, stream>>>(zIA, kHW, hdr + 8, hdr + 72, gi1, bi1, (long)kBC * 64 * kHW);
  k_gemm_w<false><<<dim3(22, 1, kBC), 128, 0, stream>>>(Wis2, zIA, zIB, kHW, kHW, sIM, sIM);
  k_bnstats<<<dim3(64), 256, 0, stream>>>(zIB, kHW, sIM, hdr + 8, hdr + 72);
  k_bnrelu<<<dim3(4200), 256, 0, stream>>>(zIB, kHW, hdr + 8, hdr + 72, gi2, bi2, (long)kBC * 64 * kHW);
  k_headscore<<<dim3(66), 256, 0, stream>>>(zIB, kHW, sIM, Wis3, scbuf);
  k_reduce<<<1, 256, 0, stream>>>(scbuf, (long)kBC * kHW, hdr + 3);

  // 8) final deterministic reductions + combine
  k_reduce<<<1, 256, 0, stream>>>(uvbuf, (long)kBC * kS * 2, hdr + 0);
  k_reduce<<<1, 256, 0, stream>>>(mskbuf, (long)kBC * kS, hdr + 1);
  k_reduce<<<1, 256, 0, stream>>>(contrib, (long)kM, hdr + 4);
  k_final<<<1, 1, 0, stream>>>(hdr, (float*)d_out);
}
